// MAF_GNN_3152505995975
// MI455X (gfx1250) — matrix operand order repaired and checked by reading, not compiled, not run on hardware
//
#include <hip/hip_runtime.h>
#include <hip/hip_bf16.h>

// ---------- types / helpers ----------
typedef __attribute__((ext_vector_type(16))) __bf16 v16bf;
typedef __attribute__((ext_vector_type(8)))  float  v8f;

__device__ __forceinline__ unsigned short f2bf(float f) {
  unsigned int u = __float_as_uint(f);
  u += 0x7fffu + ((u >> 16) & 1u);          // RNE
  return (unsigned short)(u >> 16);
}
__device__ __forceinline__ float bf2f(unsigned short h) {
  return __uint_as_float(((unsigned int)h) << 16);
}
__device__ __forceinline__ float sigmoidf_(float x) { return 1.f / (1.f + __expf(-x)); }

#define BM 128
#define BN 64
#define BK 32
#define APITCH 40   // ushort pitch (80 B) -> 16B-aligned rows, staggered banks

// ---------- Tensor Data Mover support (gfx1250) ----------
#if defined(__HIP_DEVICE_COMPILE__) && defined(__gfx1250__) && \
    __has_builtin(__builtin_amdgcn_tensor_load_to_lds) &&      \
    __has_builtin(__builtin_amdgcn_s_wait_tensorcnt)
#define USE_TDM 1
#else
#define USE_TDM 0
#endif

#if USE_TDM
typedef unsigned int tdm_v4u __attribute__((ext_vector_type(4)));
typedef int          tdm_v8i __attribute__((ext_vector_type(8)));
typedef int          tdm_v4i __attribute__((ext_vector_type(4)));

// DMA a (tile_h x tile_w) bf16 tile (row stride = stride_elems) from global
// into LDS at byte offset lds_off, padding +16B every 64B so the LDS tile has
// an 80B (APITCH) row pitch matching the WMMA fragment loader.
__device__ __forceinline__ void tdm_load_tile(unsigned lds_off,
                                              const unsigned short* gsrc,
                                              unsigned tile_w, unsigned tile_h,
                                              unsigned long long stride_elems) {
  unsigned long long ga = (unsigned long long)gsrc;
  tdm_v4u g0;
  g0[0] = 1u;                                              // count=1, user desc
  g0[1] = lds_off;                                         // lds_addr (bytes)
  g0[2] = (unsigned)ga;                                    // global_addr[31:0]
  g0[3] = (unsigned)((ga >> 32) & 0x01ffffffu) | (2u << 30); // addr[56:32] | type=2
  const unsigned w0 = (1u << 16)        // data_size = 1 -> 2-byte elements
                    | (1u << 20)        // pad_enable
                    | (3u << 22)        // pad_interval: every 16 DWORDs (64B)
                    | (3u << 25);       // pad_amount:   4 DWORDs (16B)
  const unsigned td0 = tile_w;          // tensor_dim0 (exact tile, no OOB)
  const unsigned td1 = tile_h;          // tensor_dim1
  const unsigned w1 = (td0 & 0xffffu) << 16;
  const unsigned w2 = (td0 >> 16) | ((td1 & 0xffffu) << 16);
  const unsigned w3 = (td1 >> 16) | (tile_w << 16);        // tile_dim0
  const unsigned w4 = tile_h & 0xffffu;                    // tile_dim1, tile_dim2=0
  const unsigned w5 = (unsigned)(stride_elems & 0xffffffffull);
  const unsigned w6 = (unsigned)(stride_elems >> 32);      // stride0[47:32]
  tdm_v8i g1 = { (int)w0, (int)w1, (int)w2, (int)w3,
                 (int)w4, (int)w5, (int)w6, 0 };
  tdm_v4i z4 = { 0, 0, 0, 0 };
#if __clang_major__ >= 23
  tdm_v8i z8 = { 0, 0, 0, 0, 0, 0, 0, 0 };
  __builtin_amdgcn_tensor_load_to_lds(g0, g1, z4, z4, z8, 0);
#else
  __builtin_amdgcn_tensor_load_to_lds(g0, g1, z4, z4, 0);
#endif
}
#endif  // USE_TDM

__device__ __forceinline__ unsigned lds_byte_off(const void* p) {
  return (unsigned)(unsigned long long)p;   // low 32 bits of LDS aperture addr
}

// ---------- main WMMA GEMM: C[M,Nc] = A[M,K] @ Bt[Nc,K]^T (batched) ----------
// A: f32 or bf16 row-major; Bt: bf16, row n holds column n of B (length K).
// TRANSOUT: store C^T (Nc x M) bf16; else C row-major bf16 with optional bias+relu.
// bf16-A path: double-buffered TDM global->LDS DMA; f32-A path: cooperative load.
template <bool AF32, bool TRANSOUT, bool RB>
__global__ __launch_bounds__(128) void gemm_k(
    const void* __restrict__ Av, const unsigned short* __restrict__ Bt,
    const float* __restrict__ bias, unsigned short* __restrict__ C,
    int M, int K, int Ncols,
    long long strA, long long strB, long long strC)
{
  __shared__ __align__(16) unsigned short sA[2 * BM * APITCH];
  __shared__ __align__(16) unsigned short sB[2 * BN * APITCH];

  const int tid  = threadIdx.x;
  const int lane = tid & 31;
  const int wv   = tid >> 5;
  const int l16  = lane & 15;
  const int hi   = lane >> 4;
  const int bm   = blockIdx.y * BM;
  const int bn   = blockIdx.x * BN;
  const long long bz = blockIdx.z;

  const float*          Af = (const float*)Av + bz * strA;
  const unsigned short* Ah = (const unsigned short*)Av + bz * strA;
  const unsigned short* Bb = Bt + bz * strB;

  v8f acc[2][4] = {};
  union FragU { v16bf v; uint4 q[2]; };

  // fragment-load + 8 WMMAs on one staged (A,B) tile pair
  auto compute_tiles = [&](const unsigned short* cA, const unsigned short* cB) {
    FragU a0, a1;
    {
      int row = wv * 32 + l16;
      const uint4* p = (const uint4*)&cA[row * APITCH + hi * 8];
      a0.q[0] = p[0]; a0.q[1] = p[2];            // K {0..7,16..23}/{8..15,24..31}
      row = wv * 32 + 16 + l16;
      p = (const uint4*)&cA[row * APITCH + hi * 8];
      a1.q[0] = p[0]; a1.q[1] = p[2];
    }
    #pragma unroll
    for (int ni = 0; ni < 4; ++ni) {
      FragU bfr;
      const uint4* p = (const uint4*)&cB[(ni * 16 + l16) * APITCH + hi * 8];
      bfr.q[0] = p[0]; bfr.q[1] = p[2];          // K {0..7,16..23}/{8..15,24..31}
      acc[0][ni] = __builtin_amdgcn_wmma_f32_16x16x32_bf16(
          false, a0.v, false, bfr.v, (short)0, acc[0][ni], false, false);
      acc[1][ni] = __builtin_amdgcn_wmma_f32_16x16x32_bf16(
          false, a1.v, false, bfr.v, (short)0, acc[1][ni], false, false);
    }
  };

#if USE_TDM
  if (!AF32) {
    // ---- TDM double-buffered pipeline ----
    const unsigned offA = lds_byte_off(&sA[0]);
    const unsigned offB = lds_byte_off(&sB[0]);
    const int nk = K / BK;
    if (wv == 0) {   // one DMA-issuing wave per workgroup
      tdm_load_tile(offA, Ah + (long long)bm * K, BK, BM, (unsigned long long)K);
      tdm_load_tile(offB, Bb + (long long)bn * K, BK, BN, (unsigned long long)K);
    }
    for (int kt = 0; kt < nk; ++kt) {
      if (wv == 0) __builtin_amdgcn_s_wait_tensorcnt(0);   // pair kt landed
      __syncthreads();   // consumers of buf (kt-1) done; buf kt visible to all
      if (wv == 0 && kt + 1 < nk) {
        const int nb = (kt + 1) & 1;
        tdm_load_tile(offA + nb * (BM * APITCH * 2),
                      Ah + (long long)bm * K + (long long)(kt + 1) * BK,
                      BK, BM, (unsigned long long)K);
        tdm_load_tile(offB + nb * (BN * APITCH * 2),
                      Bb + (long long)bn * K + (long long)(kt + 1) * BK,
                      BK, BN, (unsigned long long)K);
      }
      compute_tiles(&sA[(kt & 1) * BM * APITCH], &sB[(kt & 1) * BN * APITCH]);
    }
  } else
#endif
  {
    // ---- cooperative-load path (f32 A with in-flight bf16 convert) ----
    for (int k0 = 0; k0 < K; k0 += BK) {
      {
        const int r = tid;
        uint4* dst = (uint4*)&sA[r * APITCH];
        if (AF32) {
          const float* src = Af + (long long)(bm + r) * K + k0;
          #pragma unroll
          for (int q = 0; q < 4; ++q) {
            float4 f0 = ((const float4*)src)[2 * q];
            float4 f1 = ((const float4*)src)[2 * q + 1];
            union { unsigned short us[8]; uint4 u; } pk;
            pk.us[0] = f2bf(f0.x); pk.us[1] = f2bf(f0.y);
            pk.us[2] = f2bf(f0.z); pk.us[3] = f2bf(f0.w);
            pk.us[4] = f2bf(f1.x); pk.us[5] = f2bf(f1.y);
            pk.us[6] = f2bf(f1.z); pk.us[7] = f2bf(f1.w);
            dst[q] = pk.u;
          }
        } else {
          const uint4* src = (const uint4*)(Ah + (long long)(bm + r) * K + k0);
          dst[0] = src[0]; dst[1] = src[1]; dst[2] = src[2]; dst[3] = src[3];
        }
      }
      {
        const int r = tid >> 1, hf = tid & 1;
        const uint4* src = (const uint4*)(Bb + (long long)(bn + r) * K + k0 + hf * 16);
        uint4* dst = (uint4*)&sB[r * APITCH + hf * 16];
        dst[0] = src[0]; dst[1] = src[1];
      }
      __syncthreads();
      compute_tiles(sA, sB);
      __syncthreads();
    }
  }

  // --- epilogue ---
  unsigned short* Cb = C + bz * strC;
  #pragma unroll
  for (int mi = 0; mi < 2; ++mi) {
    #pragma unroll
    for (int ni = 0; ni < 4; ++ni) {
      const int n  = bn + ni * 16 + l16;
      const int m0 = bm + wv * 32 + mi * 16 + hi * 8;
      if (TRANSOUT) {
        union { unsigned short us[8]; uint4 u; } pk;
        #pragma unroll
        for (int i = 0; i < 8; ++i) pk.us[i] = f2bf(acc[mi][ni][i]);
        *(uint4*)(Cb + (long long)n * M + m0) = pk.u;      // 16B store, column of C
      } else {
        const float bv = RB ? bias[n] : 0.f;
        #pragma unroll
        for (int i = 0; i < 8; ++i) {
          float v = acc[mi][ni][i] + bv;
          if (RB) v = v > 0.f ? v : 0.f;
          Cb[(long long)(m0 + i) * Ncols + n] = f2bf(v);
        }
      }
    }
  }
}

// ---------- weight transpose + bf16 convert: out[c*R+r] = bf16(in[r*C+c]) ----------
__global__ __launch_bounds__(256) void k_wt(const float* __restrict__ in,
                                            unsigned short* __restrict__ out,
                                            int R, int Cc) {
  int idx = blockIdx.x * 256 + threadIdx.x;
  if (idx >= R * Cc) return;
  int r = idx / Cc, c = idx - r * Cc;
  out[(long long)c * R + r] = f2bf(in[idx]);
}

// ---------- 70th-percentile threshold: coarse 4096-bin histogram ----------
__global__ __launch_bounds__(256) void k_hist1(const float* __restrict__ adj,
                                               unsigned int* __restrict__ lobin,
                                               unsigned int* __restrict__ rem) {
  const int b = blockIdx.x;
  __shared__ unsigned int hist[4096];
  __shared__ unsigned int part[256];
  for (int i = threadIdx.x; i < 4096; i += 256) hist[i] = 0;
  __syncthreads();
  const float* p = adj + (long long)b * 262144;
  for (int i = threadIdx.x; i < 262144; i += 256) {
    int bin = (int)(p[i] * 4096.f);
    bin = bin < 0 ? 0 : (bin > 4095 ? 4095 : bin);
    atomicAdd(&hist[bin], 1u);
  }
  __syncthreads();
  unsigned int s = 0;
  for (int j = 0; j < 16; ++j) s += hist[threadIdx.x * 16 + j];
  part[threadIdx.x] = s;
  __syncthreads();
  if (threadIdx.x == 0) {
    const unsigned int R = 183501u;   // rank count for 70th percentile of 512*512
    unsigned int cum = 0; int seg = 0;
    for (; seg < 255; ++seg) { if (cum + part[seg] >= R) break; cum += part[seg]; }
    int bin = seg * 16;
    for (; bin < seg * 16 + 15; ++bin) { if (cum + hist[bin] >= R) break; cum += hist[bin]; }
    lobin[b] = (unsigned int)bin;
    rem[b]   = R - cum;
  }
}

// ---------- refine inside the selected coarse bin (4096 sub-bins) ----------
__global__ __launch_bounds__(256) void k_hist2(const float* __restrict__ adj,
                                               const unsigned int* __restrict__ lobin,
                                               const unsigned int* __restrict__ rem,
                                               float* __restrict__ thr) {
  const int b = blockIdx.x;
  __shared__ unsigned int hist[4096];
  __shared__ unsigned int part[256];
  for (int i = threadIdx.x; i < 4096; i += 256) hist[i] = 0;
  __syncthreads();
  const float lo = (float)lobin[b] * (1.f / 4096.f);
  const float* p = adj + (long long)b * 262144;
  for (int i = threadIdx.x; i < 262144; i += 256) {
    int bin = (int)((p[i] - lo) * 16777216.f);
    if (bin >= 0 && bin < 4096) atomicAdd(&hist[bin], 1u);
  }
  __syncthreads();
  unsigned int s = 0;
  for (int j = 0; j < 16; ++j) s += hist[threadIdx.x * 16 + j];
  part[threadIdx.x] = s;
  __syncthreads();
  if (threadIdx.x == 0) {
    const unsigned int R = rem[b];
    unsigned int cum = 0; int seg = 0;
    for (; seg < 255; ++seg) { if (cum + part[seg] >= R) break; cum += part[seg]; }
    int bin = seg * 16;
    for (; bin < seg * 16 + 15; ++bin) { if (cum + hist[bin] >= R) break; cum += hist[bin]; }
    thr[b] = lo + (float)(bin + 1) * (1.f / 16777216.f);
  }
}

// ---------- degree -> dis = rsqrt(1 + #offdiag above thr) ----------
__global__ __launch_bounds__(128) void k_deg(const float* __restrict__ adj,
                                             const float* __restrict__ thr,
                                             float* __restrict__ dis) {
  const int i = blockIdx.x, b = blockIdx.y;
  const float t = thr[b];
  const float* row = adj + ((long long)b * 512 + i) * 512;
  int c = 0;
  for (int j = threadIdx.x; j < 512; j += 128)
    c += (row[j] > t && j != i) ? 1 : 0;
  __shared__ int red[128];
  red[threadIdx.x] = c;
  __syncthreads();
  for (int s = 64; s > 0; s >>= 1) {
    if (threadIdx.x < s) red[threadIdx.x] += red[threadIdx.x + s];
    __syncthreads();
  }
  if (threadIdx.x == 0) dis[(long long)b * 512 + i] = rsqrtf((float)(red[0] + 1));
}

// ---------- build normalized adjacency in bf16 ----------
__global__ __launch_bounds__(256) void k_an(const float* __restrict__ adj,
                                            const float* __restrict__ thr,
                                            const float* __restrict__ dis,
                                            unsigned short* __restrict__ An) {
  const int i = blockIdx.x, b = blockIdx.y;
  const float t  = thr[b];
  const float di = dis[(long long)b * 512 + i];
  const float* row = adj + ((long long)b * 512 + i) * 512;
  unsigned short* orow = An + ((long long)b * 512 + i) * 512;
  for (int j = threadIdx.x; j < 512; j += 256) {
    float v = (row[j] > t || j == i) ? di * dis[(long long)b * 512 + j] : 0.f;
    orow[j] = f2bf(v);
  }
}

// ---------- mean over nodes: g[br][b][h] = mean_n H2[b][n][h] ----------
__global__ __launch_bounds__(256) void k_pool(const unsigned short* __restrict__ H2,
                                              float* __restrict__ g, int br) {
  const int b = blockIdx.x, h = threadIdx.x;
  const unsigned short* base = H2 + (long long)b * 512 * 256 + h;
  float s = 0.f;
  for (int n = 0; n < 512; ++n) s += bf2f(base[(long long)n * 256]);
  g[((long long)br * 64 + b) * 256 + h] = s * (1.f / 512.f);
}

// ---------- AAWL head -> feat[b][0:768] ----------
__global__ __launch_bounds__(256) void k_aawl(const float* __restrict__ g,
                                              const float* __restrict__ atlasW,
                                              const float* __restrict__ nodeW,
                                              const float* __restrict__ nodeB,
                                              float* __restrict__ feat) {
  const int b = blockIdx.x, h = threadIdx.x;
  __shared__ float red[256];
  __shared__ float mean3[3], max3[3], wgt[3];
  __shared__ float cm[256], cx[256];
  float gv[3], ov[3];
  #pragma unroll
  for (int i = 0; i < 3; ++i) gv[i] = g[((long long)i * 64 + b) * 256 + h];
  for (int i = 0; i < 3; ++i) {
    red[h] = gv[i]; __syncthreads();
    for (int s = 128; s > 0; s >>= 1) { if (h < s) red[h] += red[h + s]; __syncthreads(); }
    if (h == 0) mean3[i] = red[0] * (1.f / 256.f);
    __syncthreads();
    red[h] = gv[i]; __syncthreads();
    for (int s = 128; s > 0; s >>= 1) { if (h < s) red[h] = fmaxf(red[h], red[h + s]); __syncthreads(); }
    if (h == 0) max3[i] = red[0];
    __syncthreads();
  }
  if (h < 3) {
    float a = 0.f;
    for (int j = 0; j < 3; ++j) a += atlasW[h * 3 + j] * (mean3[j] + max3[j]);
    wgt[h] = sigmoidf_(a);
  }
  __syncthreads();
  float mx = -1e30f, sm = 0.f;
  for (int i = 0; i < 3; ++i) { ov[i] = gv[i] * wgt[i]; sm += ov[i]; mx = fmaxf(mx, ov[i]); }
  cm[h] = sm * (1.f / 3.f); cx[h] = mx;
  __syncthreads();
  float conv = nodeB[0];
  for (int k = -1; k <= 1; ++k) {
    int idx = h + k;
    if (idx >= 0 && idx < 256) conv += nodeW[k + 1] * cm[idx] + nodeW[3 + k + 1] * cx[idx];
  }
  const float gate = sigmoidf_(conv);
  for (int i = 0; i < 3; ++i)
    feat[(long long)b * 1536 + i * 256 + h] = gv[i] + ov[i] * gate;
}

// ---------- MHCAF head per branch -> feat[b][768 + br*256 + h] ----------
__global__ __launch_bounds__(256) void k_mhcaf(const float* __restrict__ g, int br,
                                               const float* __restrict__ Wtv,
                                               const float* __restrict__ btv,
                                               const float* __restrict__ mhaWi,
                                               const float* __restrict__ mhaBi,
                                               const float* __restrict__ Wo,
                                               const float* __restrict__ bo,
                                               const float* __restrict__ FW,
                                               const float* __restrict__ fb,
                                               float* __restrict__ feat) {
  const int b = blockIdx.x, t = threadIdx.x;
  __shared__ float sv[256], sV[256], sA2[256];
  const float* gb = g + ((long long)br * 64 + b) * 256;
  float a = btv[t];
  for (int k = 0; k < 256; ++k) a += gb[k] * Wtv[t * 256 + k];
  sv[t] = a; __syncthreads();
  a = mhaBi[512 + t];
  const float* Wv = mhaWi + 512 * 256 + t * 256;
  for (int k = 0; k < 256; ++k) a += sv[k] * Wv[k];
  sV[t] = a; __syncthreads();
  a = bo[t];
  for (int k = 0; k < 256; ++k) a += sV[k] * Wo[t * 256 + k];
  sA2[t] = a; __syncthreads();
  a = fb[t];
  const float* w = FW + (long long)t * 512;
  for (int k = 0; k < 256; ++k) a += sA2[k] * (w[k] + w[256 + k]);   // comb = [A, A]
  feat[(long long)b * 1536 + 768 + br * 256 + t] = a;
}

// ---------- final linear: out[b][o] = feat[b] . out_W[o] + out_b[o] ----------
__global__ __launch_bounds__(256) void k_final(const float* __restrict__ feat,
                                               const float* __restrict__ outW,
                                               const float* __restrict__ outB,
                                               float* __restrict__ out) {
  const int b = blockIdx.x, t = threadIdx.x;
  __shared__ float r0[256], r1[256];
  float p0 = 0.f, p1 = 0.f;
  for (int k = t; k < 1536; k += 256) {
    const float f = feat[(long long)b * 1536 + k];
    p0 += f * outW[k];
    p1 += f * outW[1536 + k];
  }
  r0[t] = p0; r1[t] = p1; __syncthreads();
  for (int s = 128; s > 0; s >>= 1) {
    if (t < s) { r0[t] += r0[t + s]; r1[t] += r1[t + s]; }
    __syncthreads();
  }
  if (t == 0) { out[b * 2 + 0] = r0[0] + outB[0]; out[b * 2 + 1] = r1[0] + outB[1]; }
}

// ---------- host ----------
extern "C" void kernel_launch(void* const* d_in, const int* in_sizes, int n_in,
                              void* d_out, int out_size, void* d_ws, size_t ws_size,
                              hipStream_t stream) {
  (void)in_sizes; (void)n_in; (void)out_size; (void)ws_size;
  const int Bb = 64, Nn = 512, Ff = 512, Hh = 256;

  // Input order: x1,adj1,x2,adj2,x3,adj3 then params flattened by sorted keys:
  // 6:atlas_W 7..12:fc3[i].{W,b} 13..24:gcn[i].{W1,W2,b1,b2} 25:mha_Wi 26:mha_Wo
  // 27:mha_bi 28:mha_bo 29:node_b 30:node_w 31:out_W 32:out_b
  // 33..38:to_k 39..44:to_q 45..50:to_v[i].{W,b}

  char* ws = (char*)d_ws;
  size_t off = 0;
  auto alloc = [&](size_t bytes) -> char* {
    char* p = ws + off; off += (bytes + 255) & ~(size_t)255; return p;
  };
  float*          thr   = (float*)alloc(Bb * 4);
  unsigned int*   lobin = (unsigned int*)alloc(Bb * 4);
  unsigned int*   rem   = (unsigned int*)alloc(Bb * 4);
  float*          dis   = (float*)alloc((size_t)Bb * Nn * 4);
  unsigned short* An    = (unsigned short*)alloc((size_t)Bb * Nn * Nn * 2);
  unsigned short* W1t   = (unsigned short*)alloc((size_t)Hh * Ff * 2);
  unsigned short* W2t   = (unsigned short*)alloc((size_t)Hh * Hh * 2);
  unsigned short* XW1t  = (unsigned short*)alloc((size_t)Bb * Hh * Nn * 2);
  unsigned short* H1    = (unsigned short*)alloc((size_t)Bb * Nn * Hh * 2);
  unsigned short* Tt    = (unsigned short*)alloc((size_t)Bb * Hh * Nn * 2);
  unsigned short* H2    = (unsigned short*)alloc((size_t)Bb * Nn * Hh * 2);
  float*          g     = (float*)alloc((size_t)3 * Bb * Hh * 4);
  float*          feat  = (float*)alloc((size_t)Bb * 6 * Hh * 4);

  const dim3 gemmGrid(Hh / BN, Nn / BM, Bb);   // (4, 4, 64)

  for (int br = 0; br < 3; ++br) {
    const float* x   = (const float*)d_in[2 * br];
    const float* adj = (const float*)d_in[2 * br + 1];
    const float* W1  = (const float*)d_in[13 + 4 * br];
    const float* W2  = (const float*)d_in[14 + 4 * br];
    const float* b1  = (const float*)d_in[15 + 4 * br];
    const float* b2  = (const float*)d_in[16 + 4 * br];

    k_wt<<<dim3((Ff * Hh + 255) / 256), 256, 0, stream>>>(W1, W1t, Ff, Hh);
    k_wt<<<dim3((Hh * Hh + 255) / 256), 256, 0, stream>>>(W2, W2t, Hh, Hh);
    k_hist1<<<Bb, 256, 0, stream>>>(adj, lobin, rem);
    k_hist2<<<Bb, 256, 0, stream>>>(adj, lobin, rem, thr);
    k_deg<<<dim3(Nn, Bb), 128, 0, stream>>>(adj, thr, dis);
    k_an<<<dim3(Nn, Bb), 256, 0, stream>>>(adj, thr, dis, An);

    // XW1^T = (x @ W1)^T  (per batch, bf16, transposed out)
    gemm_k<true, true, false><<<gemmGrid, 128, 0, stream>>>(
        x, W1t, nullptr, XW1t, Nn, Ff, Hh,
        (long long)Nn * Ff, 0LL, (long long)Hh * Nn);
    // H1 = relu(An @ XW1 + b1)
    gemm_k<false, false, true><<<gemmGrid, 128, 0, stream>>>(
        An, XW1t, b1, H1, Nn, Nn, Hh,
        (long long)Nn * Nn, (long long)Hh * Nn, (long long)Nn * Hh);
    // T^T = (H1 @ W2)^T
    gemm_k<false, true, false><<<gemmGrid, 128, 0, stream>>>(
        H1, W2t, nullptr, Tt, Nn, Hh, Hh,
        (long long)Nn * Hh, 0LL, (long long)Hh * Nn);
    // H2 = relu(An @ T + b2)
    gemm_k<false, false, true><<<gemmGrid, 128, 0, stream>>>(
        An, Tt, b2, H2, Nn, Nn, Hh,
        (long long)Nn * Nn, (long long)Hh * Nn, (long long)Nn * Hh);

    k_pool<<<Bb, 256, 0, stream>>>(H2, g, br);
  }

  k_aawl<<<Bb, 256, 0, stream>>>(g, (const float*)d_in[6], (const float*)d_in[30],
                                 (const float*)d_in[29], feat);
  for (int br = 0; br < 3; ++br) {
    k_mhcaf<<<Bb, 256, 0, stream>>>(g, br,
        (const float*)d_in[45 + 2 * br], (const float*)d_in[46 + 2 * br],
        (const float*)d_in[25], (const float*)d_in[27],
        (const float*)d_in[26], (const float*)d_in[28],
        (const float*)d_in[7 + 2 * br], (const float*)d_in[8 + 2 * br],
        feat);
  }
  k_final<<<Bb, 256, 0, stream>>>(feat, (const float*)d_in[31],
                                  (const float*)d_in[32], (float*)d_out);
}
